// DeformableDINOHead_73126113181729
// MI455X (gfx1250) — matrix operand order repaired and checked by reading, not compiled, not run on hardware
//
#include <hip/hip_runtime.h>
#include <hip/hip_bf16.h>

// ---------------------------------------------------------------------------
// Deformable DETR encoder layer for gfx1250 (MI455X), wave32 + WMMA f16.
// GEMM: 32x64 tile per wave, register double-buffered fragments.
// ---------------------------------------------------------------------------

typedef __attribute__((ext_vector_type(16))) _Float16 v16h;
typedef __attribute__((ext_vector_type(8)))  _Float16 v8h;
typedef __attribute__((ext_vector_type(8)))  float    v8f;

static constexpr int BB   = 2;
static constexpr int LEN  = 11253;            // 92^2 + 46^2 + 23^2 + 12^2
static constexpr int M2   = BB * LEN;         // 22506 rows
static constexpr int MR   = 22528;            // rounded to 32
static constexpr int DM   = 256;
static constexpr int DFFN = 1024;

// ---------------------------------------------------------------------------
// Prep: q = f16(src + pos), src_h = f16(src)
// ---------------------------------------------------------------------------
__global__ __launch_bounds__(256)
void prep_kernel(const float* __restrict__ src, const float* __restrict__ pos,
                 _Float16* __restrict__ srch, _Float16* __restrict__ qh, size_t n)
{
    size_t i = (size_t)blockIdx.x * blockDim.x + threadIdx.x;
    if (i < n) {
        float s = src[i];
        srch[i] = (_Float16)s;
        qh[i]   = (_Float16)(s + pos[i]);
    }
}

// Weight convert+transpose: W[K x N] f32 (row major) -> Wt[N x K] f16
__global__ __launch_bounds__(256)
void wcvt_kernel(const float* __restrict__ w, _Float16* __restrict__ wt, int K, int N)
{
    size_t i = (size_t)blockIdx.x * blockDim.x + threadIdx.x;
    if (i < (size_t)K * N) {
        int k = (int)(i / N);
        int n = (int)(i % N);
        wt[(size_t)n * K + k] = (_Float16)w[i];
    }
}

// ---------------------------------------------------------------------------
// Fragment loaders.
// A fragment (16x32, f16): lane<16 -> row m0+lane, K={0..7,16..23}+k0
//                          lane>=16 -> row m0+lane-16, K={8..15,24..31}+k0
// arow is pre-offset by the lane's row and the 0/8 K-skew.
// B fragment (32x16, f16): lane&15 = column, same K split as A (0/8 skew).
// ---------------------------------------------------------------------------
__device__ __forceinline__ v16h load_afrag(const _Float16* arow, int k0)
{
    v8h lo = *(const v8h*)(arow + k0);
    v8h hi = *(const v8h*)(arow + k0 + 16);
    v16h a;
#pragma unroll
    for (int i = 0; i < 8; ++i) { a[i] = lo[i]; a[i + 8] = hi[i]; }
    return a;
}

__device__ __forceinline__ v16h load_bfrag(const _Float16* bcol, int k0)
{
    v8h lo = *(const v8h*)(bcol + k0);
    v8h hi = *(const v8h*)(bcol + k0 + 16);
    v16h b;
#pragma unroll
    for (int i = 0; i < 8; ++i) { b[i] = lo[i]; b[i + 8] = hi[i]; }
    return b;
}

__device__ __forceinline__ void mma_step(const v16h& x0, const v16h& x1,
                                         const v16h& y0, const v16h& y1,
                                         const v16h& y2, const v16h& y3,
                                         v8f acc[2][4])
{
    acc[0][0] = __builtin_amdgcn_wmma_f32_16x16x32_f16(false, x0, false, y0, (short)0, acc[0][0], false, false);
    acc[1][0] = __builtin_amdgcn_wmma_f32_16x16x32_f16(false, x1, false, y0, (short)0, acc[1][0], false, false);
    acc[0][1] = __builtin_amdgcn_wmma_f32_16x16x32_f16(false, x0, false, y1, (short)0, acc[0][1], false, false);
    acc[1][1] = __builtin_amdgcn_wmma_f32_16x16x32_f16(false, x1, false, y1, (short)0, acc[1][1], false, false);
    acc[0][2] = __builtin_amdgcn_wmma_f32_16x16x32_f16(false, x0, false, y2, (short)0, acc[0][2], false, false);
    acc[1][2] = __builtin_amdgcn_wmma_f32_16x16x32_f16(false, x1, false, y2, (short)0, acc[1][2], false, false);
    acc[0][3] = __builtin_amdgcn_wmma_f32_16x16x32_f16(false, x0, false, y3, (short)0, acc[0][3], false, false);
    acc[1][3] = __builtin_amdgcn_wmma_f32_16x16x32_f16(false, x1, false, y3, (short)0, acc[1][3], false, false);
}

// ---------------------------------------------------------------------------
// WMMA GEMM: C[M x N] = A[M x K](f16, row-major) * B (as Bt[N x K], f16) + bias
// One wave -> 32x64 C tile; ping-pong double buffering of A/B fragments so
// each WMMA group only waits on loads issued one stage earlier.
// ---------------------------------------------------------------------------
template<bool RELU, bool STORE_H>
__global__ __launch_bounds__(128)
void wmma_gemm_kernel(const _Float16* __restrict__ A,
                      const _Float16* __restrict__ Bt,
                      const float*  __restrict__ bias,
                      float*        __restrict__ Cf,
                      _Float16*     __restrict__ Ch,
                      int M, int N, int K)
{
    const int wave = (int)((blockIdx.x * blockDim.x + threadIdx.x) >> 5);
    const int lane = threadIdx.x & 31;
    const int ntiles = N >> 6;               // 64-column strips
    const int mt = wave / ntiles;
    const int nt = wave - mt * ntiles;
    const int m0 = mt << 5;                  // 32 rows per wave
    const int n0 = nt << 6;
    if (m0 >= M) return;                     // wave-uniform exit (EXEC stays all-1)

    const int askew = (lane < 16) ? 0 : 8;
    int mrow0 = m0 + (lane & 15);        if (mrow0 >= M) mrow0 = M - 1;
    int mrow1 = m0 + 16 + (lane & 15);   if (mrow1 >= M) mrow1 = M - 1;
    const _Float16* arow0 = A + (size_t)mrow0 * K + askew;
    const _Float16* arow1 = A + (size_t)mrow1 * K + askew;

    const int nb  = lane & 15;
    const int bkb = (lane < 16) ? 0 : 8;
    const _Float16* bc0 = Bt + (size_t)(n0 + nb) * K + bkb;
    const _Float16* bc1 = bc0 + (size_t)16 * K;
    const _Float16* bc2 = bc0 + (size_t)32 * K;
    const _Float16* bc3 = bc0 + (size_t)48 * K;

    v8f acc[2][4] = {};

    // stage 0 fragments
    v16h a0 = load_afrag(arow0, 0);
    v16h a1 = load_afrag(arow1, 0);
    v16h b0 = load_bfrag(bc0, 0);
    v16h b1 = load_bfrag(bc1, 0);
    v16h b2 = load_bfrag(bc2, 0);
    v16h b3 = load_bfrag(bc3, 0);

    int k0 = 0;
    for (;;) {
        const int kn = k0 + 32;
        v16h na0, na1, nb0, nb1, nb2, nb3;
        if (kn < K) {                         // prefetch stage k+1 into regs
            na0 = load_afrag(arow0, kn);
            na1 = load_afrag(arow1, kn);
            nb0 = load_bfrag(bc0, kn);
            nb1 = load_bfrag(bc1, kn);
            nb2 = load_bfrag(bc2, kn);
            nb3 = load_bfrag(bc3, kn);
            if (kn + 32 < K) {
                __builtin_prefetch(arow0 + kn + 32, 0, 0);   // global_prefetch_b8
                __builtin_prefetch(arow1 + kn + 32, 0, 0);
            }
        }
        mma_step(a0, a1, b0, b1, b2, b3, acc);
        if (kn >= K) break;

        const int kn2 = kn + 32;
        if (kn2 < K) {                        // prefetch stage k+2 into regs
            a0 = load_afrag(arow0, kn2);
            a1 = load_afrag(arow1, kn2);
            b0 = load_bfrag(bc0, kn2);
            b1 = load_bfrag(bc1, kn2);
            b2 = load_bfrag(bc2, kn2);
            b3 = load_bfrag(bc3, kn2);
        }
        mma_step(na0, na1, nb0, nb1, nb2, nb3, acc);
        if (kn2 >= K) break;
        k0 = kn2;
    }

    // C layout per 16x16 tile: lane<16 -> rows +r, lane>=16 -> rows +8+r
    const int mi = (lane < 16) ? 0 : 8;
#pragma unroll
    for (int i = 0; i < 2; ++i) {
#pragma unroll
        for (int j = 0; j < 4; ++j) {
            const int nc = n0 + j * 16 + nb;
            const float bv = bias[nc];
#pragma unroll
            for (int r = 0; r < 8; ++r) {
                const int mm = m0 + i * 16 + mi + r;
                if (mm < M) {
                    float v = acc[i][j][r] + bv;
                    if (RELU) v = fmaxf(v, 0.0f);
                    if (STORE_H) Ch[(size_t)mm * N + nc] = (_Float16)v;
                    else         Cf[(size_t)mm * N + nc] = v;
                }
            }
        }
    }
}

// ---------------------------------------------------------------------------
// Deformable attention sampling. One wave per (b, q, h); 32 lanes = 32 channels
// so every value gather is one coalesced 128B line. Softmax over the 16
// (level,point) logits folded in (uniform across lanes).
// ---------------------------------------------------------------------------
__global__ __launch_bounds__(256)
void deform_sample_kernel(const float* __restrict__ value,   // (B*Len, 256) head-major
                          const float* __restrict__ offb,    // (B*Lq, 256)
                          const float* __restrict__ attnb,   // (B*Lq, 128) logits
                          const float* __restrict__ refp,    // (B*Lq, L, 2)
                          _Float16* __restrict__ sampled)    // (B*Lq, 256)
{
    const int wid  = (int)((blockIdx.x * blockDim.x + threadIdx.x) >> 5);
    const int lane = threadIdx.x & 31;
    if (wid >= M2 * 8) return;
    const int h  = wid & 7;
    const int bq = wid >> 3;
    const int b  = bq / LEN;

    // softmax over 16 logits
    const float* lgp = attnb + (size_t)bq * 128 + h * 16;
    float lg[16];
    float mx = -3.0e38f;
#pragma unroll
    for (int j = 0; j < 16; ++j) { lg[j] = lgp[j]; mx = fmaxf(mx, lg[j]); }
    float s = 0.0f;
#pragma unroll
    for (int j = 0; j < 16; ++j) { lg[j] = __expf(lg[j] - mx); s += lg[j]; }
    const float inv = 1.0f / s;

    const float* op = offb + (size_t)bq * 256 + h * 32;   // (L,P,2)
    const float* rp = refp + (size_t)bq * 8;              // (L,2)

    float acc = 0.0f;
    int start = 0;
    const int dims[4] = {92, 46, 23, 12};
#pragma unroll
    for (int l = 0; l < 4; ++l) {
        const int wl = dims[l], hl = dims[l];
        const float* vb = value + ((size_t)b * LEN + start) * 256 + h * 32 + lane;
        // x = ref_x*w + off_x - 0.5 (normalizer cancels), same for y
        const float rx = rp[l * 2 + 0] * (float)wl - 0.5f;
        const float ry = rp[l * 2 + 1] * (float)hl - 0.5f;
#pragma unroll
        for (int p = 0; p < 4; ++p) {
            const float x = rx + op[(l * 4 + p) * 2 + 0];
            const float y = ry + op[(l * 4 + p) * 2 + 1];
            const float xf = floorf(x), yf = floorf(y);
            const float lx = x - xf, ly = y - yf;
            const int x0 = (int)xf, y0 = (int)yf;
            const int x1 = x0 + 1,  y1 = y0 + 1;
            float g00 = 0.0f, g10 = 0.0f, g01 = 0.0f, g11 = 0.0f;
            if (y0 >= 0 && y0 < hl) {
                if (x0 >= 0 && x0 < wl) g00 = vb[(size_t)(y0 * wl + x0) * 256];
                if (x1 >= 0 && x1 < wl) g10 = vb[(size_t)(y0 * wl + x1) * 256];
            }
            if (y1 >= 0 && y1 < hl) {
                if (x0 >= 0 && x0 < wl) g01 = vb[(size_t)(y1 * wl + x0) * 256];
                if (x1 >= 0 && x1 < wl) g11 = vb[(size_t)(y1 * wl + x1) * 256];
            }
            const float w = lg[l * 4 + p] * inv;
            acc += w * ((1.0f - lx) * (1.0f - ly) * g00 + lx * (1.0f - ly) * g10 +
                        (1.0f - lx) * ly * g01 + lx * ly * g11);
        }
        start += wl * hl;
    }
    sampled[(size_t)bq * 256 + h * 32 + lane] = (_Float16)acc;
}

// ---------------------------------------------------------------------------
// Residual add + LayerNorm over 256. One wave per row, shfl_xor reductions.
// ---------------------------------------------------------------------------
__global__ __launch_bounds__(256)
void add_ln_kernel(const float* __restrict__ a, const float* __restrict__ b,
                   const float* __restrict__ g, const float* __restrict__ beta,
                   float* __restrict__ xf, _Float16* __restrict__ xh, int rows)
{
    const int row  = (int)((blockIdx.x * blockDim.x + threadIdx.x) >> 5);
    const int lane = threadIdx.x & 31;
    if (row >= rows) return;
    const float* ra = a + (size_t)row * 256;
    const float* rb = b + (size_t)row * 256;
    float v[8];
    float sum = 0.0f;
#pragma unroll
    for (int j = 0; j < 8; ++j) { v[j] = ra[lane + 32 * j] + rb[lane + 32 * j]; sum += v[j]; }
#pragma unroll
    for (int o = 16; o > 0; o >>= 1) sum += __shfl_xor(sum, o, 32);
    const float mean = sum * (1.0f / 256.0f);
    float var = 0.0f;
#pragma unroll
    for (int j = 0; j < 8; ++j) { const float d = v[j] - mean; var += d * d; }
#pragma unroll
    for (int o = 16; o > 0; o >>= 1) var += __shfl_xor(var, o, 32);
    const float rstd = rsqrtf(var * (1.0f / 256.0f) + 1e-5f);
#pragma unroll
    for (int j = 0; j < 8; ++j) {
        const int c = lane + 32 * j;
        const float y = g[c] * (v[j] - mean) * rstd + beta[c];
        if (xf) xf[(size_t)row * 256 + c] = y;
        if (xh) xh[(size_t)row * 256 + c] = (_Float16)y;
    }
}

// ---------------------------------------------------------------------------
extern "C" void kernel_launch(void* const* d_in, const int* in_sizes, int n_in,
                              void* d_out, int out_size, void* d_ws, size_t ws_size,
                              hipStream_t stream)
{
    (void)in_sizes; (void)n_in; (void)out_size; (void)ws_size;

    const float* src     = (const float*)d_in[0];
    const float* pos     = (const float*)d_in[1];
    const float* refp    = (const float*)d_in[2];
    const float* w_value = (const float*)d_in[3];
    const float* b_value = (const float*)d_in[4];
    const float* w_off   = (const float*)d_in[5];
    const float* b_off   = (const float*)d_in[6];
    const float* w_attn  = (const float*)d_in[7];
    const float* b_attn  = (const float*)d_in[8];
    const float* w_out   = (const float*)d_in[9];
    const float* b_out   = (const float*)d_in[10];
    const float* ln1_g   = (const float*)d_in[11];
    const float* ln1_b   = (const float*)d_in[12];
    const float* w1      = (const float*)d_in[13];
    const float* b1      = (const float*)d_in[14];
    const float* w2      = (const float*)d_in[15];
    const float* b2      = (const float*)d_in[16];
    const float* ln2_g   = (const float*)d_in[17];
    const float* ln2_b   = (const float*)d_in[18];

    // ---- carve workspace (256B aligned) ----
    char* wsp = (char*)d_ws;
    auto carve = [&](size_t bytes) -> void* {
        void* r = (void*)wsp;
        wsp += (bytes + 255) & ~(size_t)255;
        return r;
    };
    _Float16* qh      = (_Float16*)carve((size_t)MR * DM * 2);
    _Float16* srch    = (_Float16*)carve((size_t)MR * DM * 2);
    _Float16* wv_t    = (_Float16*)carve((size_t)DM * DM * 2);
    _Float16* woff_t  = (_Float16*)carve((size_t)DM * DM * 2);
    _Float16* wattn_t = (_Float16*)carve((size_t)DM * 128 * 2);
    _Float16* wout_t  = (_Float16*)carve((size_t)DM * DM * 2);
    _Float16* w1_t    = (_Float16*)carve((size_t)DM * DFFN * 2);
    _Float16* w2_t    = (_Float16*)carve((size_t)DFFN * DM * 2);
    float*    value   = (float*)carve((size_t)MR * DM * 4);
    float*    offb    = (float*)carve((size_t)MR * DM * 4);
    float*    attnb   = (float*)carve((size_t)MR * 128 * 4);
    _Float16* sampled = (_Float16*)carve((size_t)MR * DM * 2);
    float*    src2    = (float*)carve((size_t)MR * DM * 4);
    float*    xbuf    = (float*)carve((size_t)MR * DM * 4);
    _Float16* xh      = (_Float16*)carve((size_t)MR * DM * 2);
    _Float16* ffh     = (_Float16*)carve((size_t)MR * DFFN * 2);
    float*    ffn     = (float*)carve((size_t)MR * DM * 4);

    // ---- prep: f16 activations + transposed f16 weights ----
    {
        size_t n = (size_t)M2 * DM;
        prep_kernel<<<(unsigned)((n + 255) / 256), 256, 0, stream>>>(src, pos, srch, qh, n);
    }
    auto cvt = [&](const float* w, _Float16* wt, int K, int N) {
        size_t n = (size_t)K * N;
        wcvt_kernel<<<(unsigned)((n + 255) / 256), 256, 0, stream>>>(w, wt, K, N);
    };
    cvt(w_value, wv_t,    DM, DM);
    cvt(w_off,   woff_t,  DM, DM);
    cvt(w_attn,  wattn_t, DM, 128);
    cvt(w_out,   wout_t,  DM, DM);
    cvt(w1,      w1_t,    DM, DFFN);
    cvt(w2,      w2_t,    DFFN, DM);

    const int mtiles = (M2 + 31) / 32;                 // 704 (32 rows per wave)
    auto gemm_grid = [&](int N) -> unsigned {
        long waves = (long)mtiles * (N >> 6);
        return (unsigned)((waves * 32 + 127) / 128);
    };

    // ---- projections ----
    wmma_gemm_kernel<false, false><<<gemm_grid(DM), 128, 0, stream>>>(
        srch, wv_t, b_value, value, (_Float16*)nullptr, M2, DM, DM);
    wmma_gemm_kernel<false, false><<<gemm_grid(DM), 128, 0, stream>>>(
        qh, woff_t, b_off, offb, (_Float16*)nullptr, M2, DM, DM);
    wmma_gemm_kernel<false, false><<<gemm_grid(128), 128, 0, stream>>>(
        qh, wattn_t, b_attn, attnb, (_Float16*)nullptr, M2, 128, DM);

    // ---- deformable sampling (softmax fused) ----
    {
        long waves = (long)M2 * 8;
        unsigned blocks = (unsigned)((waves * 32 + 255) / 256);
        deform_sample_kernel<<<blocks, 256, 0, stream>>>(value, offb, attnb, refp, sampled);
    }

    // ---- output projection ----
    wmma_gemm_kernel<false, false><<<gemm_grid(DM), 128, 0, stream>>>(
        sampled, wout_t, b_out, src2, (_Float16*)nullptr, M2, DM, DM);

    // ---- LN1: x = LN(src + src2) ----
    {
        unsigned blocks = (unsigned)(((long)M2 * 32 + 255) / 256);
        add_ln_kernel<<<blocks, 256, 0, stream>>>(src, src2, ln1_g, ln1_b, xbuf, xh, M2);
    }

    // ---- FFN ----
    wmma_gemm_kernel<true, true><<<gemm_grid(DFFN), 128, 0, stream>>>(
        xh, w1_t, b1, (float*)nullptr, ffh, M2, DFFN, DM);
    wmma_gemm_kernel<false, false><<<gemm_grid(DM), 128, 0, stream>>>(
        ffh, w2_t, b2, ffn, (_Float16*)nullptr, M2, DM, DFFN);

    // ---- LN2 -> output ----
    {
        unsigned blocks = (unsigned)(((long)M2 * 32 + 255) / 256);
        add_ln_kernel<<<blocks, 256, 0, stream>>>(xbuf, ffn, ln2_g, ln2_b,
                                                  (float*)d_out, (_Float16*)nullptr, M2);
    }
}
